// MultiHeadAttention_773094113988
// MI455X (gfx1250) — matrix operand order repaired and checked by reading, not compiled, not run on hardware
//
#include <hip/hip_runtime.h>
#include <cstdint>

// ---------------------------------------------------------------------------
// Types for CDNA5 WMMA (gfx1250, wave32)
// ---------------------------------------------------------------------------
typedef __bf16 bf16_t;
typedef __attribute__((ext_vector_type(8)))  __bf16 v8bf;
typedef __attribute__((ext_vector_type(16))) __bf16 v16bf;
typedef __attribute__((ext_vector_type(8)))  float  v8f;

#define EMBED 768
#define HEADS 12
#define DH    64
#define BATCH 16
#define SEQ   512
#define MTOT  (BATCH * SEQ)     // 8192
#define N_QKV (3 * EMBED)       // 2304

// round-to-nearest-even f32 -> bf16 bits
__device__ __forceinline__ unsigned short f2bf(float f) {
    unsigned int u = __float_as_uint(f);
    u += 0x7FFFu + ((u >> 16) & 1u);
    return (unsigned short)(u >> 16);
}

// Build A operand: one contiguous 16-element bf16 chunk (k = 16*half + s)
__device__ __forceinline__ v16bf load_a16(const bf16_t* p0) {
    return *(const v16bf*)(p0);
}

__device__ __forceinline__ v8f wmma_bf16(v16bf a, v16bf b, v8f c) {
    return __builtin_amdgcn_wmma_f32_16x16x32_bf16(
        /*neg_a=*/false, a, /*neg_b=*/false, b,
        /*c_mod=*/(short)0, c, /*reuse_a=*/false, /*reuse_b=*/false);
}

// ---------------------------------------------------------------------------
// Conversion kernels
// ---------------------------------------------------------------------------
__global__ void cvt_bf16(const float* __restrict__ in,
                         unsigned short* __restrict__ out, int n) {
    int i = blockIdx.x * blockDim.x + threadIdx.x;
    if (i < n) out[i] = f2bf(in[i]);
}

// in [rows][cols] f32  ->  out [cols][rows] bf16 (transpose for B operand)
__global__ void cvt_bf16_t(const float* __restrict__ in,
                           unsigned short* __restrict__ out,
                           int rows, int cols) {
    int i = blockIdx.x * blockDim.x + threadIdx.x;
    if (i < rows * cols) {
        int r = i / cols, c = i % cols;
        out[(size_t)c * rows + r] = f2bf(in[i]);
    }
}

// ---------------------------------------------------------------------------
// QKV GEMM: C = x(8192x768) @ Wqkv(768x2304) + bqkv, epilogue scatters
// Q (scaled 1/sqrt(DH)) [b][h][n][64], K [b][h][n][64], V^T [b][h][64][n], bf16
// Block 256 thr = 8 waves (2 M x 4 N); 128x256 block tile; 64x64 per wave
// (4x4 WMMA tiles); K-step 32.  vmem:wmma ratio = 1.0 b128/wmma.
// ---------------------------------------------------------------------------
__global__ __launch_bounds__(256) void gemm_qkv(
    const bf16_t* __restrict__ A,    // [8192][768]
    const bf16_t* __restrict__ BT,   // [2304][768] (N-major)
    const float*  __restrict__ bias, // [2304]
    unsigned short* __restrict__ Qb,
    unsigned short* __restrict__ Kb,
    unsigned short* __restrict__ Vt) {
    const int K = EMBED;
    const int lane = threadIdx.x & 31;
    const int wid  = threadIdx.x >> 5;
    const int waveM = wid & 1, waveN = wid >> 1;
    const int half = lane >> 4, lr = lane & 15;
    const int rowBase = blockIdx.y * 128 + waveM * 64;
    const int colBase = blockIdx.x * 256 + waveN * 64;

    v8f acc[4][4];
    const v8f vzero = {0.f, 0.f, 0.f, 0.f, 0.f, 0.f, 0.f, 0.f};
#pragma unroll
    for (int mt = 0; mt < 4; ++mt)
#pragma unroll
        for (int nt = 0; nt < 4; ++nt) acc[mt][nt] = vzero;

    for (int k = 0; k < K; k += 32) {
        v16bf a[4];
#pragma unroll
        for (int mt = 0; mt < 4; ++mt) {
            int row = rowBase + mt * 16 + lr;
            a[mt] = load_a16(A + (size_t)row * K + k + half * 16);
        }
        v16bf b[4];
#pragma unroll
        for (int nt = 0; nt < 4; ++nt) {
            int col = colBase + nt * 16 + lr;
            b[nt] = *(const v16bf*)(BT + (size_t)col * K + k + half * 16);
        }
#pragma unroll
        for (int mt = 0; mt < 4; ++mt)
#pragma unroll
            for (int nt = 0; nt < 4; ++nt)
                acc[mt][nt] = wmma_bf16(a[mt], b[nt], acc[mt][nt]);
    }

#pragma unroll
    for (int mt = 0; mt < 4; ++mt) {
#pragma unroll
        for (int nt = 0; nt < 4; ++nt) {
            int col   = colBase + nt * 16 + lr;
            int which = col / EMBED;
            int rem   = col % EMBED;
            int head  = rem >> 6;
            int d     = rem & 63;
            float bv  = bias[col];
#pragma unroll
            for (int r = 0; r < 8; ++r) {
                int row = rowBase + mt * 16 + half * 8 + r;
                float v = acc[mt][nt][r] + bv;
                int bb = row >> 9, nn = row & 511;
                size_t bh = (size_t)bb * HEADS + head;
                if (which == 0)
                    Qb[(bh * SEQ + nn) * DH + d] = f2bf(v * 0.125f);
                else if (which == 1)
                    Kb[(bh * SEQ + nn) * DH + d] = f2bf(v);
                else
                    Vt[(bh * DH + d) * SEQ + nn] = f2bf(v);
            }
        }
    }
}

// ---------------------------------------------------------------------------
// Attention: one block (4 waves, 128 thr) per (b, h, 16-query-row tile)
// Phase1: scores 16x512 via WMMA -> LDS f32
// Phase2: softmax in LDS -> normalized P bf16 in LDS
// Phase3: ctx = P @ V via WMMA (V^T rows contiguous) -> bf16 ctx row-major
// ---------------------------------------------------------------------------
__global__ __launch_bounds__(128) void attn_kernel(
    const bf16_t* __restrict__ Qb,
    const bf16_t* __restrict__ Kb,
    const bf16_t* __restrict__ Vt,
    unsigned short* __restrict__ ctxb) {
    __shared__ __align__(32) float          s_sc[16 * SEQ];
    __shared__ __align__(32) unsigned short s_p[16 * SEQ];

    const int bid = blockIdx.x;
    const int qt  = bid & 31;       // query tile (16 rows)
    const int bh  = bid >> 5;       // b*HEADS + h
    const int b_  = bh / HEADS, h = bh % HEADS;

    const int lane = threadIdx.x & 31;
    const int wid  = threadIdx.x >> 5;   // 0..3
    const int half = lane >> 4, lr = lane & 15;

    const bf16_t* Qh = Qb + (size_t)bh * SEQ * DH;
    const bf16_t* Kh = Kb + (size_t)bh * SEQ * DH;
    const bf16_t* Vh = Vt + (size_t)bh * DH * SEQ;

    // ---- Phase 1: scores columns [wid*128, wid*128+128) ----
    const int qrow = qt * 16 + lr;
    v16bf aq[2];
#pragma unroll
    for (int ks = 0; ks < 2; ++ks)
        aq[ks] = load_a16(Qh + (size_t)qrow * DH + ks * 32 + half * 16);

#pragma unroll
    for (int nt = 0; nt < 8; ++nt) {
        const int kcol = wid * 128 + nt * 16 + lr;   // key row = B column
        v8f acc = {0.f, 0.f, 0.f, 0.f, 0.f, 0.f, 0.f, 0.f};
#pragma unroll
        for (int ks = 0; ks < 2; ++ks) {
            v16bf bk = *(const v16bf*)(Kh + (size_t)kcol * DH + ks * 32 + half * 16);
            acc = wmma_bf16(aq[ks], bk, acc);
        }
#pragma unroll
        for (int r = 0; r < 8; ++r)
            s_sc[(half * 8 + r) * SEQ + wid * 128 + nt * 16 + lr] = acc[r];
    }
    __syncthreads();

    // ---- Phase 2: softmax, 8 lanes per row ----
    {
        const int row = threadIdx.x >> 3;   // 0..15
        const int t   = threadIdx.x & 7;
        float m = -1e30f;
        for (int j = t; j < SEQ; j += 8) m = fmaxf(m, s_sc[row * SEQ + j]);
        m = fmaxf(m, __shfl_xor(m, 1, 32));
        m = fmaxf(m, __shfl_xor(m, 2, 32));
        m = fmaxf(m, __shfl_xor(m, 4, 32));
        float sum = 0.f;
        for (int j = t; j < SEQ; j += 8) {
            float e = __expf(s_sc[row * SEQ + j] - m);
            s_sc[row * SEQ + j] = e;
            sum += e;
        }
        sum += __shfl_xor(sum, 1, 32);
        sum += __shfl_xor(sum, 2, 32);
        sum += __shfl_xor(sum, 4, 32);
        const float inv = 1.0f / sum;
        for (int j = t; j < SEQ; j += 8)
            s_p[row * SEQ + j] = f2bf(s_sc[row * SEQ + j] * inv);
    }
    __syncthreads();

    // ---- Phase 3: ctx tile, dh columns [wid*16, wid*16+16) ----
    const bf16_t* sp = (const bf16_t*)s_p;
    v8f acc = {0.f, 0.f, 0.f, 0.f, 0.f, 0.f, 0.f, 0.f};
    for (int ks = 0; ks < SEQ / 32; ++ks) {
        v16bf a = load_a16(sp + lr * SEQ + ks * 32 + half * 16);
        v16bf b = *(const v16bf*)(Vh + (size_t)(wid * 16 + lr) * SEQ + ks * 32 + half * 16);
        acc = wmma_bf16(a, b, acc);
    }
#pragma unroll
    for (int r = 0; r < 8; ++r) {
        int row = qt * 16 + half * 8 + r;
        int col = h * DH + wid * 16 + lr;
        ctxb[((size_t)b_ * SEQ + row) * EMBED + col] = f2bf(acc[r]);
    }
}

// ---------------------------------------------------------------------------
// Output GEMM: out = ctx(8192x768) @ Wo(768x768) + bo, f32 output
// Same 64x64-per-wave tiling as gemm_qkv; block tile 128x256.
// ---------------------------------------------------------------------------
__global__ __launch_bounds__(256) void gemm_out(
    const bf16_t* __restrict__ A,    // [8192][768] ctx bf16
    const bf16_t* __restrict__ BT,   // [768][768] Wo^T bf16
    const float*  __restrict__ bias, // [768]
    float* __restrict__ C) {
    const int K = EMBED, N = EMBED;
    const int lane = threadIdx.x & 31;
    const int wid  = threadIdx.x >> 5;
    const int waveM = wid & 1, waveN = wid >> 1;
    const int half = lane >> 4, lr = lane & 15;
    const int rowBase = blockIdx.y * 128 + waveM * 64;
    const int colBase = blockIdx.x * 256 + waveN * 64;

    v8f acc[4][4];
    const v8f vzero = {0.f, 0.f, 0.f, 0.f, 0.f, 0.f, 0.f, 0.f};
#pragma unroll
    for (int mt = 0; mt < 4; ++mt)
#pragma unroll
        for (int nt = 0; nt < 4; ++nt) acc[mt][nt] = vzero;

    for (int k = 0; k < K; k += 32) {
        v16bf a[4];
#pragma unroll
        for (int mt = 0; mt < 4; ++mt) {
            int row = rowBase + mt * 16 + lr;
            a[mt] = load_a16(A + (size_t)row * K + k + half * 16);
        }
        v16bf b[4];
#pragma unroll
        for (int nt = 0; nt < 4; ++nt) {
            int col = colBase + nt * 16 + lr;
            b[nt] = *(const v16bf*)(BT + (size_t)col * K + k + half * 16);
        }
#pragma unroll
        for (int mt = 0; mt < 4; ++mt)
#pragma unroll
            for (int nt = 0; nt < 4; ++nt)
                acc[mt][nt] = wmma_bf16(a[mt], b[nt], acc[mt][nt]);
    }

#pragma unroll
    for (int mt = 0; mt < 4; ++mt) {
#pragma unroll
        for (int nt = 0; nt < 4; ++nt) {
            int col  = colBase + nt * 16 + lr;
            float bv = bias[col];
#pragma unroll
            for (int r = 0; r < 8; ++r) {
                int row = rowBase + mt * 16 + half * 8 + r;
                C[(size_t)row * N + col] = acc[mt][nt][r] + bv;
            }
        }
    }
}

// ---------------------------------------------------------------------------
// Host launcher
// ---------------------------------------------------------------------------
extern "C" void kernel_launch(void* const* d_in, const int* in_sizes, int n_in,
                              void* d_out, int out_size, void* d_ws, size_t ws_size,
                              hipStream_t stream) {
    const float* x    = (const float*)d_in[0];
    const float* Wqkv = (const float*)d_in[1];
    const float* bqkv = (const float*)d_in[2];
    const float* Wo   = (const float*)d_in[3];
    const float* bo   = (const float*)d_in[4];
    float* out = (float*)d_out;

    char* ws = (char*)d_ws;
    size_t off = 0;
    auto alloc = [&](size_t bytes) -> char* {
        char* p = ws + off;
        off += (bytes + 255) & ~(size_t)255;
        return p;
    };
    unsigned short* xb    = (unsigned short*)alloc((size_t)MTOT * EMBED * 2);
    unsigned short* wqkvT = (unsigned short*)alloc((size_t)N_QKV * EMBED * 2);
    unsigned short* woT   = (unsigned short*)alloc((size_t)EMBED * EMBED * 2);
    unsigned short* Qb    = (unsigned short*)alloc((size_t)BATCH * HEADS * SEQ * DH * 2);
    unsigned short* Kb    = (unsigned short*)alloc((size_t)BATCH * HEADS * SEQ * DH * 2);
    unsigned short* Vt    = (unsigned short*)alloc((size_t)BATCH * HEADS * SEQ * DH * 2);
    unsigned short* ctxb  = (unsigned short*)alloc((size_t)MTOT * EMBED * 2);

    // 1) precision conversion + weight transposes (B operands want N-major)
    {
        int n = MTOT * EMBED;
        cvt_bf16<<<(n + 255) / 256, 256, 0, stream>>>(x, xb, n);
    }
    {
        int n = EMBED * N_QKV;
        cvt_bf16_t<<<(n + 255) / 256, 256, 0, stream>>>(Wqkv, wqkvT, EMBED, N_QKV);
    }
    {
        int n = EMBED * EMBED;
        cvt_bf16_t<<<(n + 255) / 256, 256, 0, stream>>>(Wo, woT, EMBED, EMBED);
    }

    // 2) QKV projection with fused Q-scale + head scatter (V transposed)
    gemm_qkv<<<dim3(N_QKV / 256, MTOT / 128), 256, 0, stream>>>(
        (const bf16_t*)xb, (const bf16_t*)wqkvT, bqkv, Qb, Kb, Vt);

    // 3) attention: 16 query rows per block, full softmax in LDS
    attn_kernel<<<BATCH * HEADS * (SEQ / 16), 128, 0, stream>>>(
        (const bf16_t*)Qb, (const bf16_t*)Kb, (const bf16_t*)Vt, ctxb);

    // 4) output projection, fp32 result
    gemm_out<<<dim3(EMBED / 256, MTOT / 128), 256, 0, stream>>>(
        (const bf16_t*)ctxb, (const bf16_t*)woT, bo, out);
}
